// PointTokenizer_7181185319612
// MI455X (gfx1250) — matrix operand order repaired and checked by reading, not compiled, not run on hardware
//
#include <hip/hip_runtime.h>
#include <hip/hip_bf16.h>

#define BB  16
#define NN  8192
#define GG  512
#define MM  32
#define ENC 384
#define EPSV 1e-5f

typedef _Float16 half16 __attribute__((ext_vector_type(16)));
typedef float    f32x8  __attribute__((ext_vector_type(8)));
typedef unsigned int v4u __attribute__((ext_vector_type(4)));
typedef int      v4i   __attribute__((ext_vector_type(4)));
typedef int      v8i   __attribute__((ext_vector_type(8)));

#if defined(__has_builtin)
#if __has_builtin(__builtin_amdgcn_tensor_load_to_lds)
#define HAVE_TDM 1
#endif
#endif
#ifndef HAVE_TDM
#define HAVE_TDM 0
#endif

__device__ inline f32x8 zero8() {
  f32x8 v = {0.f,0.f,0.f,0.f,0.f,0.f,0.f,0.f};
  return v;
}

// ---- WMMA fragment loaders (CDNA5 16-bit layouts, 16x16x32 f16) ----
// A (16x32): lanes 0-15 -> M=0..15; lanes 16-31 same M, K offset +8.
__device__ inline half16 load_a_frag(const _Float16* base, int rowBase, int stride, int k0) {
  int lane = threadIdx.x & 31;
  int hi   = lane >> 4;
  int row  = rowBase + (lane & 15);
  const unsigned* pu = (const unsigned*)(base + row * stride + k0);
  union { unsigned u[8]; half16 h; } f;
#pragma unroll
  for (int v = 0; v < 4; ++v) f.u[v] = pu[hi * 4 + v];
#pragma unroll
  for (int v = 0; v < 4; ++v) f.u[4 + v] = pu[8 + hi * 4 + v];
  return f.h;
}

// B (32x16): lane = N column; lanes 0-15 hold K=0..7,16..23, lanes 16-31 K=8..15,24..31.
// panelT stored [n][32] halves so K pairs are contiguous dwords.
__device__ inline half16 load_b_frag(const _Float16* panelT, int nBase) {
  int lane = threadIdx.x & 31;
  int hi   = lane >> 4;
  int n    = nBase + (lane & 15);
  const unsigned* pu = (const unsigned*)(panelT + n * 32);
  union { unsigned u[8]; half16 h; } f;
#pragma unroll
  for (int v = 0; v < 8; ++v) f.u[v] = pu[(v >> 2) * 8 + hi * 4 + (v & 3)];
  return f.h;
}

// -------- K-panel staging: WT is pre-transposed f16 [N][K] (K contiguous) --------
// Panel in LDS: [n][32] halves = one 64B row per n -> a strided 2D tile.

#if HAVE_TDM
// Issue a 2D TDM tile load: tile_dim0=32 elems (x), tile_dim1=ncols rows (y),
// row stride = Kfull elems, data_size = 2 bytes. Wave 0 only (TDM ignores EXEC).
__device__ inline void tdm_issue(const _Float16* WT, unsigned ldsAddr,
                                 int ncols, int Kfull, int k0) {
  if ((threadIdx.x >> 5) == 0) {
    unsigned long long ga = (unsigned long long)(uintptr_t)(WT + k0);
    v4u g0;
    g0[0] = 1u;                                   // count=1, user descriptor
    g0[1] = ldsAddr;                              // lds_addr [63:32]
    g0[2] = (unsigned)(ga & 0xffffffffu);         // global_addr [95:64]
    g0[3] = (unsigned)((ga >> 32) & 0x01ffffffu)  // global_addr [120:96]
            | (2u << 30);                         // type=2 ("image")
    v8i g1;
    g1[0] = (int)(1u << 16);                      // data_size=1 (2 bytes)
    g1[1] = (int)(32u << 16);                     // tensor_dim0[15:0]=32 @ [63:48]
    g1[2] = (int)(((unsigned)ncols & 0xffffu) << 16); // tensor_dim1[15:0] @ [95:80]
    g1[3] = (int)(32u << 16);                     // tile_dim0=32 @ [127:112]
    g1[4] = (int)((unsigned)ncols & 0xffffu);     // tile_dim1=ncols, tile_dim2=0
    g1[5] = (int)Kfull;                           // tensor_dim0_stride[31:0]
    g1[6] = 0;
    g1[7] = 0;
    v4i z4 = {0, 0, 0, 0};
#if __clang_major__ >= 23
    v8i z8 = {0, 0, 0, 0, 0, 0, 0, 0};
    __builtin_amdgcn_tensor_load_to_lds(g0, g1, z4, z4, z8, 0);
#else
    __builtin_amdgcn_tensor_load_to_lds(g0, g1, z4, z4, 0);
#endif
  }
}
__device__ inline void panel_issue(_Float16* panelT, const _Float16* WT,
                                   int ncols, int Kfull, int k0) {
  tdm_issue(WT, (unsigned)(uintptr_t)panelT, ncols, Kfull, k0);
}
__device__ inline void panel_wait_1() {
  if ((threadIdx.x >> 5) == 0) __builtin_amdgcn_s_wait_tensorcnt(1);
}
__device__ inline void panel_wait_0() {
  if ((threadIdx.x >> 5) == 0) __builtin_amdgcn_s_wait_tensorcnt(0);
}
#else
// Fallback: cooperative 128-bit copies (4x uint4 per n row).
__device__ inline void panel_issue(_Float16* panelT, const _Float16* WT,
                                   int ncols, int Kfull, int k0) {
  for (int idx = threadIdx.x; idx < ncols * 4; idx += 256) {
    int n = idx >> 2, c = idx & 3;
    ((uint4*)(panelT + n * 32))[c] =
        ((const uint4*)(WT + (size_t)n * Kfull + k0))[c];
  }
}
__device__ inline void panel_wait_1() {}
__device__ inline void panel_wait_0() {}
#endif

// -------------------- weight f32 [K,N] -> f16 transposed [N,K] --------------------
__global__ void convertT_kernel(const float* __restrict__ src, _Float16* __restrict__ dst,
                                int K, int Ncols) {
  int i = blockIdx.x * blockDim.x + threadIdx.x;
  if (i < K * Ncols) {
    int k = i / Ncols, n = i % Ncols;   // coalesced reads
    dst[(size_t)n * K + k] = (_Float16)src[i];
  }
}

// -------------------- farthest point sampling --------------------
__global__ void __launch_bounds__(512) fps_kernel(const float* __restrict__ pts,
                                                  float* __restrict__ centers) {
  __shared__ float sval[512];
  __shared__ int   sidx[512];
  __shared__ float sc[3];
  int b = blockIdx.x, t = threadIdx.x;
  const float* P = pts + (size_t)b * NN * 3;
  float dist[NN / 512];
#pragma unroll
  for (int i = 0; i < NN / 512; ++i) dist[i] = 1e10f;
  int far = 0;
  for (int s = 0; s < GG; ++s) {
    if (t == 0) {
      float cx = P[far * 3 + 0], cy = P[far * 3 + 1], cz = P[far * 3 + 2];
      sc[0] = cx; sc[1] = cy; sc[2] = cz;
      float* C = centers + ((size_t)b * GG + s) * 3;
      C[0] = cx; C[1] = cy; C[2] = cz;
    }
    __syncthreads();
    float cx = sc[0], cy = sc[1], cz = sc[2];
    float best = -1.f; int bestn = 0;
#pragma unroll
    for (int i = 0; i < NN / 512; ++i) {
      int n = t + i * 512;
      float dx = P[n * 3 + 0] - cx, dy = P[n * 3 + 1] - cy, dz = P[n * 3 + 2] - cz;
      float d = dx * dx + dy * dy + dz * dz;
      dist[i] = fminf(dist[i], d);
      if (dist[i] > best) { best = dist[i]; bestn = n; }
    }
    sval[t] = best; sidx[t] = bestn;
    __syncthreads();
    for (int off = 256; off > 0; off >>= 1) {
      if (t < off) {
        if (sval[t + off] > sval[t] ||
            (sval[t + off] == sval[t] && sidx[t + off] < sidx[t])) {
          sval[t] = sval[t + off]; sidx[t] = sidx[t + off];
        }
      }
      __syncthreads();
    }
    far = sidx[0];
    __syncthreads();
  }
}

// -------------------- kNN gather (top-32 nearest, ascending) --------------------
__global__ void __launch_bounds__(256) knn_kernel(const float* __restrict__ pts,
                                                  const float* __restrict__ centers,
                                                  float* __restrict__ nbhd) {
  __shared__ float sval[256];
  __shared__ int   sidx[256];
  __shared__ float c3[3];
  int bid = blockIdx.x;
  int b = bid >> 9;
  int t = threadIdx.x;
  const float* P = pts + (size_t)b * NN * 3;
  if (t < 3) c3[t] = centers[(size_t)bid * 3 + t];
  __syncthreads();
  float cx = c3[0], cy = c3[1], cz = c3[2];
  float d[NN / 256];
#pragma unroll
  for (int j = 0; j < NN / 256; ++j) {
    int n = t + j * 256;
    float dx = P[n * 3 + 0] - cx, dy = P[n * 3 + 1] - cy, dz = P[n * 3 + 2] - cz;
    d[j] = dx * dx + dy * dy + dz * dz;
  }
  for (int m = 0; m < MM; ++m) {
    float best = 3.0e38f; int bestn = 0x7fffffff;
#pragma unroll
    for (int j = 0; j < NN / 256; ++j) {
      if (d[j] < best) { best = d[j]; bestn = t + j * 256; }
    }
    sval[t] = best; sidx[t] = bestn;
    __syncthreads();
    for (int off = 128; off > 0; off >>= 1) {
      if (t < off) {
        if (sval[t + off] < sval[t] ||
            (sval[t + off] == sval[t] && sidx[t + off] < sidx[t])) {
          sval[t] = sval[t + off]; sidx[t] = sidx[t + off];
        }
      }
      __syncthreads();
    }
    int nwin = sidx[0];
    if (t == (nwin & 255)) d[nwin >> 8] = 3.4e38f;
    if (t < 3) nbhd[((size_t)bid * MM + m) * 3 + t] = P[nwin * 3 + t] - c3[t];
    __syncthreads();
  }
}

// -------------------- fused per-group encoder (WMMA + TDM panels) --------------------
// Dynamic LDS, 147840 B. F4 (f32 32x384 = 48K) reuses the X1/X2/FG region
// (lifetimes disjoint). Two panel buffers for TDM double buffering.
#define X1_OFF     0        // f16 32x128   (8K)
#define X2_OFF     8192     // f16 32x512   (32K)
#define FG_OFF     40960    // f32 256      (1K)
#define F4_OFF     0        // f32 32x384   (48K) [after stage 3]
#define X3_OFF     49152    // f16 32x512   (32K)
#define PANA_OFF   81920    // f16 <=512x32 (32K)
#define PANB_OFF   114688   // f16 <=512x32 (32K)
#define NB_OFF     147456   // f32 32x3     (384B)
#define SMEM_BYTES 147840

__global__ void __launch_bounds__(256) encoder_kernel(
    const float* __restrict__ nbhd,
    const float* __restrict__ W1, const float* __restrict__ b1,
    const float* __restrict__ g1, const float* __restrict__ be1,
    const float* __restrict__ m1, const float* __restrict__ v1,
    const _Float16* __restrict__ W2T, const float* __restrict__ b2,
    const _Float16* __restrict__ W3T, const float* __restrict__ b3,
    const float* __restrict__ g2, const float* __restrict__ be2,
    const float* __restrict__ m2, const float* __restrict__ v2,
    const _Float16* __restrict__ W4T, const float* __restrict__ b4,
    float* __restrict__ xout) {
  extern __shared__ char smem[];
  _Float16* x1   = (_Float16*)(smem + X1_OFF);
  _Float16* x2   = (_Float16*)(smem + X2_OFF);
  _Float16* x3   = (_Float16*)(smem + X3_OFF);
  _Float16* panA = (_Float16*)(smem + PANA_OFF);
  _Float16* panB = (_Float16*)(smem + PANB_OFF);
  float*    fg   = (float*)(smem + FG_OFF);
  float*    f4   = (float*)(smem + F4_OFF);
  float*    nb   = (float*)(smem + NB_OFF);

  int bid  = blockIdx.x;
  int t    = threadIdx.x;
  int wave = t >> 5;
  int lane = t & 31, hi = lane >> 4, nloc = lane & 15;
  int rt   = wave & 1;

  if (t < 96) nb[t] = nbhd[(size_t)bid * 96 + t];
  __syncthreads();

  // stage 1: 3 -> 128, BN(eval), ReLU   (VALU; K=3 too small for WMMA)
  for (int o = t; o < 32 * 128; o += 256) {
    int r = o >> 7, c = o & 127;
    float v = nb[r * 3 + 0] * W1[c] + nb[r * 3 + 1] * W1[128 + c] +
              nb[r * 3 + 2] * W1[256 + c] + b1[c];
    v = (v - m1[c]) * rsqrtf(v1[c] + EPSV) * g1[c] + be1[c];
    x1[r * 128 + c] = (_Float16)fmaxf(v, 0.f);
  }
  __syncthreads();

  // stage 2: [32,128] x [128,256] + b2 -> x2[:,256:512]
  {
    f32x8 acc[4];
#pragma unroll
    for (int c = 0; c < 4; ++c) acc[c] = zero8();
    int ctBase = (wave >> 1) * 4;
    panel_issue(panA, W2T, 256, 128, 0);
    for (int ks = 0; ks < 4; ++ks) {
      _Float16* cur = (ks & 1) ? panB : panA;
      _Float16* nxt = (ks & 1) ? panA : panB;
      if (ks + 1 < 4) { panel_issue(nxt, W2T, 256, 128, (ks + 1) * 32); panel_wait_1(); }
      else            { panel_wait_0(); }
      __syncthreads();
      half16 a = load_a_frag(x1, rt * 16, 128, ks * 32);
#pragma unroll
      for (int c = 0; c < 4; ++c) {
        half16 bfr = load_b_frag(cur, (ctBase + c) * 16);
        acc[c] = __builtin_amdgcn_wmma_f32_16x16x32_f16(false, a, false, bfr,
                                                        (short)0, acc[c], false, false);
      }
      __syncthreads();
    }
#pragma unroll
    for (int c = 0; c < 4; ++c) {
      int n = (ctBase + c) * 16 + nloc;
      float bias = b2[n];
#pragma unroll
      for (int i = 0; i < 8; ++i) {
        int row = rt * 16 + hi * 8 + i;
        x2[row * 512 + 256 + n] = (_Float16)(acc[c][i] + bias);
      }
    }
  }
  __syncthreads();

  // fg = max over 32 rows; broadcast into x2[:,0:256] (the concat)
  {
    float mx = -3.4e38f;
    for (int r = 0; r < 32; ++r) mx = fmaxf(mx, (float)x2[r * 512 + 256 + t]);
    fg[t] = mx;
    __syncthreads();
    _Float16 fgv = (_Float16)fg[t];
    for (int r = 0; r < 32; ++r) x2[r * 512 + t] = fgv;
  }
  __syncthreads();

  // stage 3: [32,512] x [512,512] + b3, BN, ReLU -> x3
  {
    f32x8 acc[8];
#pragma unroll
    for (int c = 0; c < 8; ++c) acc[c] = zero8();
    int ctBase = (wave >> 1) * 8;
    panel_issue(panA, W3T, 512, 512, 0);
    for (int ks = 0; ks < 16; ++ks) {
      _Float16* cur = (ks & 1) ? panB : panA;
      _Float16* nxt = (ks & 1) ? panA : panB;
      if (ks + 1 < 16) { panel_issue(nxt, W3T, 512, 512, (ks + 1) * 32); panel_wait_1(); }
      else             { panel_wait_0(); }
      __syncthreads();
      half16 a = load_a_frag(x2, rt * 16, 512, ks * 32);
#pragma unroll
      for (int c = 0; c < 8; ++c) {
        half16 bfr = load_b_frag(cur, (ctBase + c) * 16);
        acc[c] = __builtin_amdgcn_wmma_f32_16x16x32_f16(false, a, false, bfr,
                                                        (short)0, acc[c], false, false);
      }
      __syncthreads();
    }
#pragma unroll
    for (int c = 0; c < 8; ++c) {
      int n = (ctBase + c) * 16 + nloc;
      float bias = b3[n];
      float sc = rsqrtf(v2[n] + EPSV) * g2[n];
      float mu = m2[n], bt = be2[n];
#pragma unroll
      for (int i = 0; i < 8; ++i) {
        int row = rt * 16 + hi * 8 + i;
        float v = acc[c][i] + bias;
        v = (v - mu) * sc + bt;
        x3[row * 512 + n] = (_Float16)fmaxf(v, 0.f);
      }
    }
  }
  __syncthreads();

  // stage 4: [32,512] x [512,384] + b4 -> f4, then max over M -> x output
  {
    f32x8 acc[6];
#pragma unroll
    for (int c = 0; c < 6; ++c) acc[c] = zero8();
    int ctBase = (wave >> 1) * 6;
    panel_issue(panA, W4T, 384, 512, 0);
    for (int ks = 0; ks < 16; ++ks) {
      _Float16* cur = (ks & 1) ? panB : panA;
      _Float16* nxt = (ks & 1) ? panA : panB;
      if (ks + 1 < 16) { panel_issue(nxt, W4T, 384, 512, (ks + 1) * 32); panel_wait_1(); }
      else             { panel_wait_0(); }
      __syncthreads();
      half16 a = load_a_frag(x3, rt * 16, 512, ks * 32);
#pragma unroll
      for (int c = 0; c < 6; ++c) {
        half16 bfr = load_b_frag(cur, (ctBase + c) * 16);
        acc[c] = __builtin_amdgcn_wmma_f32_16x16x32_f16(false, a, false, bfr,
                                                        (short)0, acc[c], false, false);
      }
      __syncthreads();
    }
#pragma unroll
    for (int c = 0; c < 6; ++c) {
      int n = (ctBase + c) * 16 + nloc;
      float bias = b4[n];
#pragma unroll
      for (int i = 0; i < 8; ++i) {
        int row = rt * 16 + hi * 8 + i;
        f4[row * 384 + n] = acc[c][i] + bias;
      }
    }
  }
  __syncthreads();

  for (int n = t; n < ENC; n += 256) {
    float mx = -3.4e38f;
    for (int r = 0; r < 32; ++r) mx = fmaxf(mx, f4[r * 384 + n]);
    xout[(size_t)bid * ENC + n] = mx;
  }
}

// -------------------- pos-embed MLP on centers --------------------
__global__ void __launch_bounds__(128) pos_kernel(const float* __restrict__ centers,
    const float* __restrict__ Wp1, const float* __restrict__ bp1,
    const float* __restrict__ Wp2, const float* __restrict__ bp2,
    float* __restrict__ out) {
  __shared__ float hid[128];
  int bid = blockIdx.x;
  int b = bid >> 9, g = bid & 511;
  int h = threadIdx.x;
  float cx = centers[(size_t)bid * 3 + 0];
  float cy = centers[(size_t)bid * 3 + 1];
  float cz = centers[(size_t)bid * 3 + 2];
  float s = cx * Wp1[h] + cy * Wp1[128 + h] + cz * Wp1[256 + h] + bp1[h];
  float s3 = s * s * s;
  hid[h] = 0.5f * s * (1.f + tanhf(0.7978845608f * (s + 0.044715f * s3)));  // tanh GELU
  __syncthreads();
  size_t base = (size_t)BB * GG * ENC + ((size_t)b * (GG + 1) + (g + 1)) * ENC;
  for (int n = h; n < ENC; n += 128) {
    float acc = bp2[n];
#pragma unroll 4
    for (int k = 0; k < 128; ++k) acc += hid[k] * Wp2[k * ENC + n];
    out[base + n] = acc;
  }
}

__global__ void cls_kernel(const float* __restrict__ cls_pos, float* __restrict__ out) {
  int b = blockIdx.x, n = threadIdx.x;
  out[(size_t)BB * GG * ENC + (size_t)b * (GG + 1) * ENC + n] = cls_pos[n];
}

// -------------------- launch --------------------
extern "C" void kernel_launch(void* const* d_in, const int* in_sizes, int n_in,
                              void* d_out, int out_size, void* d_ws, size_t ws_size,
                              hipStream_t stream) {
  (void)in_sizes; (void)n_in; (void)out_size; (void)ws_size;
  const float* pts = (const float*)d_in[0];
  const float* W1  = (const float*)d_in[1];
  const float* b1  = (const float*)d_in[2];
  const float* g1  = (const float*)d_in[3];
  const float* be1 = (const float*)d_in[4];
  const float* m1  = (const float*)d_in[5];
  const float* v1  = (const float*)d_in[6];
  const float* W2  = (const float*)d_in[7];
  const float* b2  = (const float*)d_in[8];
  const float* W3  = (const float*)d_in[9];
  const float* b3  = (const float*)d_in[10];
  const float* g2  = (const float*)d_in[11];
  const float* be2 = (const float*)d_in[12];
  const float* m2  = (const float*)d_in[13];
  const float* v2  = (const float*)d_in[14];
  const float* W4  = (const float*)d_in[15];
  const float* b4  = (const float*)d_in[16];
  const float* Wp1 = (const float*)d_in[17];
  const float* bp1 = (const float*)d_in[18];
  const float* Wp2 = (const float*)d_in[19];
  const float* bp2 = (const float*)d_in[20];
  const float* cls = (const float*)d_in[21];

  char* ws = (char*)d_ws;
  float*    centers = (float*)(ws + 0);              // 16*512*3*4      = 98304
  float*    nbhd    = (float*)(ws + 98304);          // 16*512*32*3*4   = 3145728
  _Float16* W2T     = (_Float16*)(ws + 3244032);     // [256][128] f16  = 65536
  _Float16* W3T     = (_Float16*)(ws + 3309568);     // [512][512] f16  = 524288
  _Float16* W4T     = (_Float16*)(ws + 3833856);     // [384][512] f16  = 393216

  convertT_kernel<<<128, 256, 0, stream>>>(W2, W2T, 128, 256);
  convertT_kernel<<<1024, 256, 0, stream>>>(W3, W3T, 512, 512);
  convertT_kernel<<<768, 256, 0, stream>>>(W4, W4T, 512, 384);

  fps_kernel<<<BB, 512, 0, stream>>>(pts, centers);
  knn_kernel<<<BB * GG, 256, 0, stream>>>(pts, centers, nbhd);

  encoder_kernel<<<BB * GG, 256, SMEM_BYTES, stream>>>(
      nbhd, W1, b1, g1, be1, m1, v1,
      W2T, b2, W3T, b3, g2, be2, m2, v2, W4T, b4,
      (float*)d_out);

  pos_kernel<<<BB * GG, 128, 0, stream>>>(centers, Wp1, bp1, Wp2, bp2, (float*)d_out);
  cls_kernel<<<BB, ENC, 0, stream>>>(cls, (float*)d_out);
}
